// SwinTransformerBlock_9277129359822
// MI455X (gfx1250) — matrix operand order repaired and checked by reading, not compiled, not run on hardware
//
#include <hip/hip_runtime.h>
#include <cstdint>
#include <cstddef>

// ---------------------------------------------------------------------------
// CDNA5 (gfx1250) Swin transformer block. All GEMMs via v_wmma_f32_16x16x32_f16.
// ---------------------------------------------------------------------------

typedef __attribute__((ext_vector_type(16))) _Float16 v16h;
typedef __attribute__((ext_vector_type(8)))  _Float16 v8h;
typedef __attribute__((ext_vector_type(8)))  float    v8f;
typedef __attribute__((ext_vector_type(4)))  float    v4f;

#define DEVFN static __device__ __forceinline__

DEVFN v8f wmma_f16(v16h a, v16h b, v8f c) {
  // D = A(16x32 f16) * B(32x16 f16) + C(16x16 f32)
  return __builtin_amdgcn_wmma_f32_16x16x32_f16(false, a, false, b,
                                                (short)0, c, false, false);
}

// ----- problem constants (match reference) ---------------------------------
constexpr int T_    = 5;
constexpr int WSH   = 8, WSW = 8;
constexpr int SH_   = 4;              // shift
constexpr int HEADS_= 6;
constexpr int DIMC  = 192;
constexpr int HDIM  = 32;
constexpr int HIMG  = 64, WIMG = 64;
constexpr int BATCH = 2;
constexpr int NTOK  = T_ * WSH * WSW;            // 320 tokens / window
constexpr int MROWS = BATCH * T_ * HIMG * WIMG;  // 40960 token rows
constexpr float SCALE_ = 0.17677669529663687f;   // 32^-0.5

// ===========================================================================
// LayerNorm (fp32 in) -> f16 out.  One wave32 per row, C = 192 = 32*6.
// ===========================================================================
__global__ __launch_bounds__(256)
void ln_kernel(const float* __restrict__ x, const float* __restrict__ g,
               const float* __restrict__ b, _Float16* __restrict__ out, int M) {
  const int lane = threadIdx.x & 31;
  const int wave = threadIdx.x >> 5;
  const int row  = blockIdx.x * 8 + wave;
  if (row >= M) return;
  const float* xr = x + (size_t)row * DIMC;
  float v[6];
  float s = 0.f;
#pragma unroll
  for (int i = 0; i < 6; ++i) { v[i] = xr[lane + 32 * i]; s += v[i]; }
#pragma unroll
  for (int off = 16; off > 0; off >>= 1) s += __shfl_xor(s, off, 32);
  const float mu = s * (1.0f / DIMC);
  float s2 = 0.f;
#pragma unroll
  for (int i = 0; i < 6; ++i) { float d = v[i] - mu; s2 += d * d; }
#pragma unroll
  for (int off = 16; off > 0; off >>= 1) s2 += __shfl_xor(s2, off, 32);
  const float rstd = rsqrtf(s2 * (1.0f / DIMC) + 1e-5f);
  _Float16* orow = out + (size_t)row * DIMC;
#pragma unroll
  for (int i = 0; i < 6; ++i) {
    const int c = lane + 32 * i;
    orow[c] = (_Float16)((v[i] - mu) * rstd * g[c] + b[c]);
  }
}

// ===========================================================================
// WMMA GEMM:  out[M,N] = A[M,K](f16) @ W[N,K]^T(f32->f16) + bias (+ epilogue)
// Block tile 256x64x32, 256 threads = 8 waves stacked along M; each wave owns
// a 32x64 tile = 2(M) x 4(N) WMMA accumulators -> 8 WMMAs per wave per K-step
// against 6 fragment loads (2 A + 4 B).
// EPI: 0 = bias, f16 out | 1 = bias+GELU(exact), f16 out | 2 = bias+res, f32
// ===========================================================================
constexpr int BM = 256, BN = 64, BK = 32, LDT = 40; // LDT: padded LDS stride

template <int EPI>
__global__ __launch_bounds__(256)
void gemm_ws(const _Float16* __restrict__ A, const float* __restrict__ W,
             const float* __restrict__ bias, const float* __restrict__ res,
             void* __restrict__ outp, int M, int N, int K) {
  __shared__ _Float16 sA[BM * LDT];   // 20480 B
  __shared__ _Float16 sB[BN * LDT];   //  5120 B

  const int tid  = threadIdx.x;
  const int lane = tid & 31;
  const int wave = tid >> 5;            // 0..7, M sub-tile
  const int m0   = blockIdx.y * BM;
  const int n0   = blockIdx.x * BN;

  const int l15 = lane & 15;
  const int kbA = (lane < 16) ? 0 : 8;   // A-frag K base (ISA 7.12.2)
  const int kbB = (lane < 16) ? 0 : 8;   // B-frag K base
  const int brow = tid >> 2, bcol = (tid & 3) * 8;  // W tile: 64 rows x 32 f32

  v8f acc[2][4] = {};

  for (int k0 = 0; k0 < K; k0 += BK) {
    // A tile: one 64B row per thread (256 rows x 32 halves)
    const _Float16* ag = A + (size_t)(m0 + tid) * K + k0;
    v8h a0 = *(const v8h*)(ag);
    v8h a1 = *(const v8h*)(ag + 8);
    v8h a2 = *(const v8h*)(ag + 16);
    v8h a3 = *(const v8h*)(ag + 24);
    _Float16* sa = &sA[tid * LDT];
    *(v8h*)(sa)      = a0;
    *(v8h*)(sa + 8)  = a1;
    *(v8h*)(sa + 16) = a2;
    *(v8h*)(sa + 24) = a3;

    // W tile: f32 -> f16 on the fly
    const float* wg = W + (size_t)(n0 + brow) * K + k0 + bcol;
    v4f w0 = *(const v4f*)(wg);
    v4f w1 = *(const v4f*)(wg + 4);
    v8h wh;
#pragma unroll
    for (int t = 0; t < 4; ++t) { wh[t] = (_Float16)w0[t]; wh[t + 4] = (_Float16)w1[t]; }
    *(v8h*)(&sB[brow * LDT + bcol]) = wh;

    if (k0 + BK < K) {                       // global_prefetch_b8 of next slab
      __builtin_prefetch(ag + BK, 0, 0);
      __builtin_prefetch(wg + BK, 0, 0);
    }
    __syncthreads();

    v16h af[2], bf[4];
#pragma unroll
    for (int i = 0; i < 2; ++i) {
      const _Float16* p = &sA[(wave * 32 + i * 16 + l15) * LDT];
      v8h lo = *(const v8h*)(p + kbA);
      v8h hi = *(const v8h*)(p + 16 + kbA);
#pragma unroll
      for (int t = 0; t < 8; ++t) { af[i][t] = lo[t]; af[i][t + 8] = hi[t]; }
    }
#pragma unroll
    for (int j = 0; j < 4; ++j) {
      const _Float16* p = &sB[(j * 16 + l15) * LDT + kbB];
      v8h lo = *(const v8h*)(p);
      v8h hi = *(const v8h*)(p + 16);
#pragma unroll
      for (int t = 0; t < 8; ++t) { bf[j][t] = lo[t]; bf[j][t + 8] = hi[t]; }
    }
#pragma unroll
    for (int i = 0; i < 2; ++i)
#pragma unroll
      for (int j = 0; j < 4; ++j)
        acc[i][j] = wmma_f16(af[i], bf[j], acc[i][j]);
    __syncthreads();
  }

  // epilogue (C/D layout: VGPR v -> M = v (+8 for lanes 16..31), N = lane&15)
  const int mofs = (lane < 16) ? 0 : 8;
#pragma unroll
  for (int i = 0; i < 2; ++i) {
#pragma unroll
    for (int j = 0; j < 4; ++j) {
      const int n  = n0 + j * 16 + l15;
      const float bv = bias[n];
#pragma unroll
      for (int v = 0; v < 8; ++v) {
        const int m = m0 + wave * 32 + i * 16 + v + mofs;
        float val = acc[i][j][v] + bv;
        if (EPI == 1) val = 0.5f * val * (1.0f + erff(val * 0.70710678118654752f));
        if (EPI == 2) {
          val += res[(size_t)m * N + n];
          ((float*)outp)[(size_t)m * N + n] = val;
        } else {
          ((_Float16*)outp)[(size_t)m * N + n] = (_Float16)val;
        }
      }
    }
  }
}

// ===========================================================================
// Windowed attention.  grid = (5 row-tiles of 64, 128 windows, 6 heads).
// Shift-roll + window partition via index math.  S tile (64x320 fp32) plus a
// f16 probability tile live in dynamic LDS (174 KB -- CDNA5's 320KB/WGP).
// ===========================================================================
constexpr int LQS = 40;    // sQ  64 x 32 halves, stride 40
constexpr int LKS = 40;    // sK  320 x 32 halves
constexpr int LVS = 336;   // sVt 32 x 320 halves (transposed), stride 336
constexpr int LSS = 328;   // sS  64 x 320 floats / sP halves, stride 328
constexpr int OFF_K  = 64 * LQS;                 // halves
constexpr int OFF_VT = OFF_K + 320 * LKS;        // halves
constexpr int OFF_S  = (OFF_VT + 32 * LVS) * 2;  // bytes (16B aligned)
constexpr int OFF_P  = OFF_S + 64 * LSS * 4;     // bytes (16B aligned)
constexpr int ATTN_LDS_BYTES = OFF_P + 64 * LSS * 2;   // 178176 B

DEVFN int tok2row(int bimg, int wh, int ww, int n) {
  const int t  = n >> 6;
  const int p  = n & 63;
  const int ph = p >> 3, pw = p & 7;
  const int hsrc = (wh * 8 + ph + SH_) & (HIMG - 1);   // inverse of roll(-4)
  const int wsrc = (ww * 8 + pw + SH_) & (WIMG - 1);
  return ((bimg * T_ + t) * HIMG + hsrc) * WIMG + wsrc;
}

DEVFN int region_id(int r, int c) {
  const int ih = (r < HIMG - WSH) ? 0 : ((r < HIMG - SH_) ? 1 : 2);
  const int iw = (c < WIMG - WSW) ? 0 : ((c < WIMG - SH_) ? 1 : 2);
  return ih * 3 + iw;
}

__global__ __launch_bounds__(256)
void attn_kernel(const _Float16* __restrict__ qkv,
                 const float* __restrict__ bias_table,
                 _Float16* __restrict__ out) {
  extern __shared__ char smem[];
  _Float16* sQ  = (_Float16*)smem;
  _Float16* sK  = (_Float16*)smem + OFF_K;
  _Float16* sVt = (_Float16*)smem + OFF_VT;
  float*    sS  = (float*)(smem + OFF_S);
  _Float16* sP  = (_Float16*)(smem + OFF_P);

  const int rt  = blockIdx.x;          // 0..4   row tile (64 queries)
  const int b_  = blockIdx.y;          // 0..127 window
  const int hh  = blockIdx.z;          // 0..5   head
  const int bimg = b_ >> 6;
  const int win  = b_ & 63;
  const int wh = win >> 3, ww = win & 7;

  const int tid  = threadIdx.x;
  const int lane = tid & 31;
  const int wave = tid >> 5;
  const int l15  = lane & 15;

  // ---- stage Q (64x32), K (320x32), V^T (32x320) in LDS ------------------
  {
    const int r  = tid >> 2;
    const int c8 = (tid & 3) * 8;
    {
      const int row = tok2row(bimg, wh, ww, rt * 64 + r);
      v8h q = *(const v8h*)(qkv + (size_t)row * 576 + hh * HDIM + c8);
      *(v8h*)(&sQ[r * LQS + c8]) = q;
    }
#pragma unroll
    for (int i = 0; i < 5; ++i) {
      const int n   = r + i * 64;                // 0..319
      const int row = tok2row(bimg, wh, ww, n);
      v8h kk = *(const v8h*)(qkv + (size_t)row * 576 + DIMC + hh * HDIM + c8);
      *(v8h*)(&sK[n * LKS + c8]) = kk;
      v8h vv = *(const v8h*)(qkv + (size_t)row * 576 + 2 * DIMC + hh * HDIM + c8);
#pragma unroll
      for (int t = 0; t < 8; ++t) sVt[(c8 + t) * LVS + n] = vv[t];
    }
  }
  __syncthreads();

  // ---- S = scale*Q.K^T + rel_bias + mask  (4 m-tiles x 20 n-tiles) -------
  {
    const int mt = wave >> 1;                 // 0..3
    const int jh = wave & 1;                  // n tiles 0..9 or 10..19
    const int kbA = (lane < 16) ? 0 : 8;
    const int kbB = (lane < 16) ? 0 : 8;
    const int mofs = (lane < 16) ? 0 : 8;
    v16h aQ;
    {
      const _Float16* p = &sQ[(mt * 16 + l15) * LQS];
      v8h lo = *(const v8h*)(p + kbA);
      v8h hi = *(const v8h*)(p + 16 + kbA);
#pragma unroll
      for (int t = 0; t < 8; ++t) { aQ[t] = lo[t]; aQ[t + 8] = hi[t]; }
    }
    // hoist query-token decomposition (fixed across the 10 n-tiles)
    int ti[8], hi_[8], wi_[8], idi[8];
#pragma unroll
    for (int v = 0; v < 8; ++v) {
      const int itok = rt * 64 + mt * 16 + v + mofs;
      ti[v]  = itok >> 6;
      const int pi = itok & 63;
      hi_[v] = pi >> 3;
      wi_[v] = pi & 7;
      idi[v] = region_id(wh * 8 + hi_[v], ww * 8 + wi_[v]);
    }
#pragma unroll
    for (int jt = 0; jt < 10; ++jt) {
      const int nt = jh * 10 + jt;            // 0..19
      v16h bK;
      {
        const _Float16* p = &sK[(nt * 16 + l15) * LKS + kbB];
        v8h lo = *(const v8h*)(p);
        v8h hi = *(const v8h*)(p + 16);
#pragma unroll
        for (int t = 0; t < 8; ++t) { bK[t] = lo[t]; bK[t + 8] = hi[t]; }
      }
      v8f c = {};
      c = wmma_f16(aQ, bK, c);

      const int jtok = nt * 16 + l15;         // key token 0..319
      const int tj = jtok >> 6;
      const int pj = jtok & 63;
      const int hj = pj >> 3, wj = pj & 7;
      const int idj = region_id(wh * 8 + hj, ww * 8 + wj);
#pragma unroll
      for (int v = 0; v < 8; ++v) {
        const int bidx = (ti[v] - tj + 7) * 15 + (hi_[v] - hj + 7) + (wi_[v] - wj);
        float val = c[v] * SCALE_ + bias_table[bidx * HEADS_ + hh];
        if (idi[v] != idj) val -= 100.0f;
        sS[(mt * 16 + v + mofs) * LSS + jtok] = val;
      }
    }
  }
  __syncthreads();

  // ---- softmax over 320 cols; writes normalized probs as f16 into sP -----
  {
#pragma unroll
    for (int rr = 0; rr < 8; ++rr) {
      const int row = wave * 8 + rr;
      const float* sr = sS + row * LSS;
      _Float16*   pr = sP + row * LSS;
      float xv[10];
      float mx = -3.0e38f;
#pragma unroll
      for (int k = 0; k < 10; ++k) { xv[k] = sr[lane + 32 * k]; mx = fmaxf(mx, xv[k]); }
#pragma unroll
      for (int off = 16; off > 0; off >>= 1) mx = fmaxf(mx, __shfl_xor(mx, off, 32));
      float sum = 0.f;
#pragma unroll
      for (int k = 0; k < 10; ++k) { xv[k] = expf(xv[k] - mx); sum += xv[k]; }
#pragma unroll
      for (int off = 16; off > 0; off >>= 1) sum += __shfl_xor(sum, off, 32);
      const float inv = 1.0f / sum;
#pragma unroll
      for (int k = 0; k < 10; ++k) pr[lane + 32 * k] = (_Float16)(xv[k] * inv);
    }
  }
  __syncthreads();

  // ---- O = P @ V  (each wave: one (m-tile, d-tile); K = 320 in 10 steps) -
  {
    const int mt = wave >> 1;
    const int dt = wave & 1;
    const int kbA = (lane < 16) ? 0 : 8;
    const int kbB = (lane < 16) ? 0 : 8;
    const int dcol = dt * 16 + l15;           // 0..31 within head
    v8f o = {};
#pragma unroll
    for (int nc = 0; nc < 10; ++nc) {
      v16h aP;
      {
        const _Float16* p = sP + (mt * 16 + l15) * LSS + nc * 32;
        v8h lo = *(const v8h*)(p + kbA);
        v8h hi = *(const v8h*)(p + 16 + kbA);
#pragma unroll
        for (int t = 0; t < 8; ++t) { aP[t] = lo[t]; aP[t + 8] = hi[t]; }
      }
      v16h bV;
      {
        const _Float16* vp = &sVt[dcol * LVS + nc * 32 + kbB];
        v8h lo = *(const v8h*)(vp);
        v8h hi = *(const v8h*)(vp + 16);
#pragma unroll
        for (int t = 0; t < 8; ++t) { bV[t] = lo[t]; bV[t + 8] = hi[t]; }
      }
      o = wmma_f16(aP, bV, o);
    }
    const int mofs = (lane < 16) ? 0 : 8;
#pragma unroll
    for (int v = 0; v < 8; ++v) {
      const int itok = rt * 64 + mt * 16 + v + mofs;
      const int row  = tok2row(bimg, wh, ww, itok);   // inverse roll/rev
      out[(size_t)row * DIMC + hh * HDIM + dcol] = (_Float16)o[v];
    }
  }
}

// ===========================================================================
// Host-side orchestration
// ===========================================================================
extern "C" void kernel_launch(void* const* d_in, const int* in_sizes, int n_in,
                              void* d_out, int out_size, void* d_ws, size_t ws_size,
                              hipStream_t stream) {
  (void)in_sizes; (void)n_in; (void)out_size; (void)ws_size;
  const float* x      = (const float*)d_in[0];
  const float* ln1_g  = (const float*)d_in[1];
  const float* ln1_b  = (const float*)d_in[2];
  const float* qkv_w  = (const float*)d_in[3];
  const float* qkv_b  = (const float*)d_in[4];
  const float* bias_t = (const float*)d_in[5];
  const float* proj_w = (const float*)d_in[6];
  const float* proj_b = (const float*)d_in[7];
  const float* ln2_g  = (const float*)d_in[8];
  const float* ln2_b  = (const float*)d_in[9];
  const float* fc1_w  = (const float*)d_in[10];
  const float* fc1_b  = (const float*)d_in[11];
  const float* fc2_w  = (const float*)d_in[12];
  const float* fc2_b  = (const float*)d_in[13];

  char* ws = (char*)d_ws;
  size_t off = 0;
  auto carve = [&](size_t bytes) -> char* {
    char* p = ws + off;
    off += (bytes + 255) & ~(size_t)255;
    return p;
  };
  _Float16* ln1_h  = (_Float16*)carve((size_t)MROWS * DIMC * 2);       // 15.7 MB
  _Float16* qkv_h  = (_Float16*)carve((size_t)MROWS * 3 * DIMC * 2);   // 47.2 MB
  _Float16* attn_h = (_Float16*)carve((size_t)MROWS * DIMC * 2);       // 15.7 MB
  float*    x2     = (float*)   carve((size_t)MROWS * DIMC * 4);       // 31.5 MB
  _Float16* ln2_h  = (_Float16*)carve((size_t)MROWS * DIMC * 2);       // 15.7 MB
  _Float16* mlp_h  = (_Float16*)carve((size_t)MROWS * 4 * DIMC * 2);   // 62.9 MB

  const dim3 blk(256);

  // 1) LN1 -> f16
  ln_kernel<<<dim3(MROWS / 8), blk, 0, stream>>>(x, ln1_g, ln1_b, ln1_h, MROWS);
  // 2) QKV GEMM (+bias) -> f16
  gemm_ws<0><<<dim3((3 * DIMC) / BN, MROWS / BM), blk, 0, stream>>>(
      ln1_h, qkv_w, qkv_b, nullptr, qkv_h, MROWS, 3 * DIMC, DIMC);
  // 3) windowed attention (dynamic LDS 174 KB -- CDNA5 320KB/WGP)
  (void)hipFuncSetAttribute((const void*)attn_kernel,
                            hipFuncAttributeMaxDynamicSharedMemorySize,
                            ATTN_LDS_BYTES);
  attn_kernel<<<dim3(NTOK / 64, BATCH * 64, HEADS_), blk, ATTN_LDS_BYTES, stream>>>(
      qkv_h, bias_t, attn_h);
  // 4) proj GEMM + bias + residual(x) -> x2 (f32)
  gemm_ws<2><<<dim3(DIMC / BN, MROWS / BM), blk, 0, stream>>>(
      attn_h, proj_w, proj_b, x, x2, MROWS, DIMC, DIMC);
  // 5) LN2 -> f16
  ln_kernel<<<dim3(MROWS / 8), blk, 0, stream>>>(x2, ln2_g, ln2_b, ln2_h, MROWS);
  // 6) fc1 GEMM + bias + exact GELU -> f16
  gemm_ws<1><<<dim3((4 * DIMC) / BN, MROWS / BM), blk, 0, stream>>>(
      ln2_h, fc1_w, fc1_b, nullptr, mlp_h, MROWS, 4 * DIMC, DIMC);
  // 7) fc2 GEMM + bias + residual(x2) -> d_out (f32)
  gemm_ws<2><<<dim3(DIMC / BN, MROWS / BM), blk, 0, stream>>>(
      mlp_h, fc2_w, fc2_b, x2, (float*)d_out, MROWS, DIMC, 4 * DIMC);
}
